// SparseSSM_59828894433828
// MI455X (gfx1250) — compile-verified, matrix operand order checked by reading, not run on hardware
//
#include <hip/hip_runtime.h>
#include <hip/hip_bf16.h>
#include <math.h>

// ---------------------------------------------------------------------------
// Model dims (from reference)
// ---------------------------------------------------------------------------
constexpr int cV = 32000, cD = 1024, cN = 256, cF = 4096, cL = 8;
constexpr int cB = 2, cS = 2048, cT = cB * cS;   // 4096 tokens
constexpr int c3N = 3 * cN;                      // fused dt|B|C projection width
constexpr float cEPS = 1e-5f;

typedef __bf16 bf16_t;
typedef __attribute__((ext_vector_type(16))) __bf16 v16bf;
typedef __attribute__((ext_vector_type(8)))  __bf16 v8bf;
typedef __attribute__((ext_vector_type(8)))  float  v8f;

__device__ __forceinline__ v16bf frag_combine(v8bf lo, v8bf hi) {
  return __builtin_shufflevector(lo, hi, 0,1,2,3,4,5,6,7,8,9,10,11,12,13,14,15);
}

__device__ __forceinline__ float gelu_tanh(float x) {
  float x3 = x * x * x;
  return 0.5f * x * (1.0f + tanhf(0.7978845608028654f * (x + 0.044715f * x3)));
}
__device__ __forceinline__ float softplus_f(float x) {
  return (x > 20.0f) ? x : log1pf(expf(x));
}

// ---------------------------------------------------------------------------
// CDNA5 async global->LDS copy (ASYNCcnt-tracked) + wait, via inline asm
// ---------------------------------------------------------------------------
__device__ __forceinline__ unsigned lds_off(const void* p) {
  // addrspacecast generic -> LDS(3); ptrtoint of an AS3 pointer is the LDS byte offset
  return (unsigned)(unsigned long long)(__attribute__((address_space(3))) const char*)p;
}

__device__ __forceinline__ void async_copy_b128(unsigned lds_addr, const void* gaddr) {
  asm volatile("global_load_async_to_lds_b128 %0, %1, off"
               :: "v"(lds_addr), "v"((unsigned long long)(size_t)gaddr)
               : "memory");
}

template<int N>
__device__ __forceinline__ void wait_async() {
  asm volatile("s_wait_asynccnt %0" :: "n"(N) : "memory");
}

// ---------------------------------------------------------------------------
// Embedding gather: x[t, :] = emb[ids[t], :]
// ---------------------------------------------------------------------------
__global__ void k_gather(const int* __restrict__ ids, const float* __restrict__ emb,
                         float* __restrict__ x) {
  int t = blockIdx.x;
  const float* src = emb + (size_t)ids[t] * cD;
  float* dst = x + (size_t)t * cD;
  for (int i = threadIdx.x; i < cD; i += blockDim.x) dst[i] = src[i];
}

// ---------------------------------------------------------------------------
// f32 -> bf16 convert / transpose-convert (weights prepared once per launch)
// ---------------------------------------------------------------------------
__global__ void k_to_bf16(const float* __restrict__ src, bf16_t* __restrict__ dst, size_t n) {
  size_t i = (size_t)blockIdx.x * blockDim.x + threadIdx.x;
  if (i < n) dst[i] = (bf16_t)src[i];
}

// src: R x C row-major f32  ->  dst: C x R row-major bf16
__global__ void k_transpose_bf16(const float* __restrict__ src, bf16_t* __restrict__ dst,
                                 int R, int C) {
  size_t i = (size_t)blockIdx.x * blockDim.x + threadIdx.x;
  size_t n = (size_t)R * C;
  if (i >= n) return;
  int r = (int)(i / C), c = (int)(i % C);
  dst[(size_t)c * R + r] = (bf16_t)src[i];
}

// ---------------------------------------------------------------------------
// LayerNorm over D=1024, one block (256 thr) per token.
// MODE bit0: write f32, bit1: write bf16
// ---------------------------------------------------------------------------
template<int MODE>
__global__ void k_layernorm(const float* __restrict__ x, const float* __restrict__ sc,
                            const float* __restrict__ bi, float* __restrict__ yf,
                            bf16_t* __restrict__ ybf) {
  __shared__ float ssum[256], ssq[256];
  int t = blockIdx.x, tid = threadIdx.x;
  const float* row = x + (size_t)t * cD;
  float v[4], lsum = 0.f, lsq = 0.f;
#pragma unroll
  for (int i = 0; i < 4; ++i) {
    float u = row[tid + i * 256];
    v[i] = u; lsum += u; lsq += u * u;
  }
  ssum[tid] = lsum; ssq[tid] = lsq;
  __syncthreads();
  for (int off = 128; off > 0; off >>= 1) {
    if (tid < off) { ssum[tid] += ssum[tid + off]; ssq[tid] += ssq[tid + off]; }
    __syncthreads();
  }
  float mean = ssum[0] * (1.0f / cD);
  float var  = ssq[0] * (1.0f / cD) - mean * mean;
  float rstd = rsqrtf(var + cEPS);
#pragma unroll
  for (int i = 0; i < 4; ++i) {
    int d = tid + i * 256;
    float o = (v[i] - mean) * rstd * sc[d] + bi[d];
    if (MODE & 1) yf[(size_t)t * cD + d] = o;
    if (MODE & 2) ybf[(size_t)t * cD + d] = (bf16_t)o;
  }
}

// ---------------------------------------------------------------------------
// WMMA bf16 GEMM, NT layout, async-LDS double-buffered pipeline:
//   C[M x Nc] = A[M x K] (row-major bf16) * Bt[Nc x K]^T (row-major bf16)
// Block: 256 thr = 8 waves, block tile 64(M) x 128(N), K-step 32.
// Steady state (last K-step peeled, branch-free): issue 3 async b128 copies
// per wave into buffer[next], s_wait_asynccnt 3 (drains buffer[cur]), barrier,
// 2x2 WMMA tiles from LDS, barrier.  LDS rows padded to 40 elems (80B stride
// -> bank stride 20, 16 distinct banks; keeps 16B alignment for b128).
// ---------------------------------------------------------------------------
enum { EPI_NONE = 0, EPI_BIAS = 1, EPI_BIAS_GELU_BF = 2, EPI_SSM = 3, EPI_BIAS_RES = 4 };

constexpr int LPAD = 40;                        // padded row stride (elements)
constexpr unsigned ABUF = 64  * LPAD * 2;       // A buffer bytes (5120)
constexpr unsigned BBUF = 128 * LPAD * 2;       // B buffer bytes (10240)

template<int EPI>
__global__ __launch_bounds__(256) void k_gemm_bf16_nt(
    const bf16_t* __restrict__ A, const bf16_t* __restrict__ Bt,
    float* __restrict__ C, bf16_t* __restrict__ Cbf,
    int M, int Nc, int K,
    const float* __restrict__ bias, const float* __restrict__ res,
    const float* __restrict__ aux, const float* __restrict__ dskip) {
  __shared__ __align__(16) bf16_t As[2][64  * LPAD];
  __shared__ __align__(16) bf16_t Bs[2][128 * LPAD];

  const int tid  = threadIdx.x;
  const int lane = tid & 31;
  const int wave = tid >> 5;
  const int wm = wave >> 2, wn = wave & 3;
  const int rl = lane & 15;          // row (A) / col (B) inside 16-tile
  const int kg = lane >> 4;          // K-group select
  const int mBase = blockIdx.y * 64;
  const int nBase = blockIdx.x * 128;

  // ---- async copy task assignment (16B chunks, 4 chunks per 32-elem row) ----
  const int crow = tid >> 2, cchk = tid & 3;
  const bf16_t* gA  = A  + (size_t)(mBase + crow)      * K + cchk * 8;
  const bf16_t* gB0 = Bt + (size_t)(nBase + crow)      * K + cchk * 8;
  const bf16_t* gB1 = Bt + (size_t)(nBase + 64 + crow) * K + cchk * 8;

  const unsigned la  = lds_off(&As[0][0]) + (unsigned)(crow * (LPAD * 2) + cchk * 16);
  const unsigned lb0 = lds_off(&Bs[0][0]) + (unsigned)(crow * (LPAD * 2) + cchk * 16);
  const unsigned lb1 = lb0 + 64 * (LPAD * 2);

  v8f acc[2][2] = {};
  const int nk = K >> 5;

  auto compute = [&](int cur) {
    const bf16_t* a = As[cur];
    const bf16_t* b = Bs[cur];
    v16bf fa[2], fb[2];
#pragma unroll
    for (int s = 0; s < 2; ++s) {
      const int ar = wm * 32 + s * 16 + rl;
      fa[s] = frag_combine(*(const v8bf*)(a + ar * LPAD + kg * 8),
                           *(const v8bf*)(a + ar * LPAD + 16 + kg * 8));
      const int br = wn * 32 + s * 16 + rl;
      fb[s] = frag_combine(*(const v8bf*)(b + br * LPAD + kg * 8),
                           *(const v8bf*)(b + br * LPAD + 16 + kg * 8));
    }
    acc[0][0] = __builtin_amdgcn_wmma_f32_16x16x32_bf16(false, fa[0], false, fb[0],
                                                        (short)0, acc[0][0], false, false);
    acc[0][1] = __builtin_amdgcn_wmma_f32_16x16x32_bf16(false, fa[0], false, fb[1],
                                                        (short)0, acc[0][1], false, false);
    acc[1][0] = __builtin_amdgcn_wmma_f32_16x16x32_bf16(false, fa[1], false, fb[0],
                                                        (short)0, acc[1][0], false, false);
    acc[1][1] = __builtin_amdgcn_wmma_f32_16x16x32_bf16(false, fa[1], false, fb[1],
                                                        (short)0, acc[1][1], false, false);
  };

  // prologue: fill buffer 0 with K-step 0
  async_copy_b128(la,  gA);
  async_copy_b128(lb0, gB0);
  async_copy_b128(lb1, gB1);

  // steady state: branch-free issue/wait/compute
  int i = 0;
  for (; i < nk - 1; ++i) {
    const int cur = i & 1;
    const unsigned nb = (unsigned)(cur ^ 1);
    const int k = (i + 1) << 5;
    async_copy_b128(la  + nb * ABUF, gA  + k);
    async_copy_b128(lb0 + nb * BBUF, gB0 + k);
    async_copy_b128(lb1 + nb * BBUF, gB1 + k);
    wait_async<3>();       // buffer[cur] copies complete; newest 3 in flight
    __syncthreads();
    compute(cur);
    __syncthreads();       // all waves done reading buffer[cur] before reuse
  }
  // epilogue K-step: drain and compute last buffer
  wait_async<0>();
  __syncthreads();
  compute(i & 1);

  // Epilogue. C/D layout: VGPR r, lanes 0-15 -> M=r, N=lane; lanes 16-31 -> M=r+8.
  const int m0 = mBase + wm * 32;
  const int n0 = nBase + wn * 32;
#pragma unroll
  for (int ti = 0; ti < 2; ++ti) {
#pragma unroll
    for (int tj = 0; tj < 2; ++tj) {
      const int rowb = m0 + ti * 16 + 8 * kg;
      const int col  = n0 + tj * 16 + rl;
#pragma unroll
      for (int r = 0; r < 8; ++r) {
        size_t o = (size_t)(rowb + r) * Nc + col;
        float v = acc[ti][tj][r];
        if constexpr (EPI == EPI_BIAS) {
          C[o] = v + bias[col];
        } else if constexpr (EPI == EPI_BIAS_GELU_BF) {
          Cbf[o] = (bf16_t)gelu_tanh(v + bias[col]);
        } else if constexpr (EPI == EPI_SSM) {
          C[o] = v + aux[o] * dskip[col] + res[o];    // y + h*D_skip + residual
        } else if constexpr (EPI == EPI_BIAS_RES) {
          C[o] = v + bias[col] + res[o];
        } else {
          C[o] = v;
        }
      }
    }
  }
}

// ---------------------------------------------------------------------------
// SSM pointwise on fused (T x 3N) projection buffer [dt | B | C]:
//   dt = softplus(raw + b_dt); decay = exp(dt*A) (in place over dt slice);
//   binp = dt * B (in place over B slice); C slice untouched.
// ---------------------------------------------------------------------------
__global__ void k_ssm_elem(float* __restrict__ dbc,
                           const float* __restrict__ bdt, const float* __restrict__ alog) {
  size_t i = (size_t)blockIdx.x * blockDim.x + threadIdx.x;
  if (i >= (size_t)cT * cN) return;
  int t = (int)(i / cN), n = (int)(i % cN);
  size_t o = (size_t)t * c3N + n;
  float dt = softplus_f(dbc[o] + bdt[n]);
  float A = -expf(alog[n]);
  dbc[o]      = expf(dt * A);       // decay
  dbc[o + cN] = dt * dbc[o + cN];   // binp
}

// ---------------------------------------------------------------------------
// Sequential selective scan: 512 independent (b,n) recurrences over S, reading
// the fused (T x 3N) buffer; writes prod = bf16(h_s * C_s) in (T x N) layout.
// ---------------------------------------------------------------------------
__global__ void k_ssm_scan(const float* __restrict__ dbc, bf16_t* __restrict__ prod) {
  int idx = blockIdx.x * blockDim.x + threadIdx.x;   // 0 .. cB*cN-1
  int b = idx / cN, n = idx % cN;
  size_t base = (size_t)b * cS * c3N + n;
  float h = 0.f;
  for (int s = 0; s < cS; ++s) {
    size_t o = base + (size_t)s * c3N;
    h = fmaf(dbc[o], h, dbc[o + cN]);                       // decay*h + binp
    prod[(size_t)(b * cS + s) * cN + n] = (bf16_t)(h * dbc[o + 2 * cN]);
  }
}

// ---------------------------------------------------------------------------
// Host-side orchestration
// ---------------------------------------------------------------------------
extern "C" void kernel_launch(void* const* d_in, const int* in_sizes, int n_in,
                              void* d_out, int out_size, void* d_ws, size_t ws_size,
                              hipStream_t stream) {
  (void)in_sizes; (void)n_in; (void)out_size; (void)ws_size;

  const int*   ids   = (const int*)  d_in[0];
  const float* emb   = (const float*)d_in[1];
  const float* A_log = (const float*)d_in[2];
  const float* W_dt  = (const float*)d_in[3];
  const float* b_dt  = (const float*)d_in[4];
  const float* W_B   = (const float*)d_in[5];
  const float* W_C   = (const float*)d_in[6];
  const float* W_out = (const float*)d_in[7];
  const float* Dskip = (const float*)d_in[8];
  const float* ln1s  = (const float*)d_in[9];
  const float* ln1b  = (const float*)d_in[10];
  const float* ln2s  = (const float*)d_in[11];
  const float* ln2b  = (const float*)d_in[12];
  const float* W1    = (const float*)d_in[13];
  const float* b1    = (const float*)d_in[14];
  const float* W2    = (const float*)d_in[15];
  const float* b2    = (const float*)d_in[16];
  const float* lnfs  = (const float*)d_in[17];
  const float* lnfb  = (const float*)d_in[18];

  // ---- workspace carve ----
  uint8_t* wp = (uint8_t*)d_ws;
  auto carve = [&](size_t bytes) -> void* {
    void* q = wp; wp += (bytes + 255) & ~(size_t)255; return q;
  };
  bf16_t* emb_bf = (bf16_t*)carve((size_t)cV * cD * 2);
  bf16_t* WcatT  = (bf16_t*)carve((size_t)cL * c3N * cD * 2); // (3N x D) per layer
  bf16_t* WoutT  = (bf16_t*)carve((size_t)cL * cD * cN * 2);  // (D x N) per layer
  bf16_t* W1T    = (bf16_t*)carve((size_t)cL * cF * cD * 2);  // (F x D) per layer
  bf16_t* W2T    = (bf16_t*)carve((size_t)cL * cD * cF * 2);  // (D x F) per layer

  float*  x    = (float*) carve((size_t)cT * cD * 4);
  float*  h    = (float*) carve((size_t)cT * cD * 4);
  bf16_t* hbf  = (bf16_t*)carve((size_t)cT * cD * 2);
  float*  dbc  = (float*) carve((size_t)cT * c3N * 4);  // fused [dt|B|C] (T x 768)
  bf16_t* prod = (bf16_t*)carve((size_t)cT * cN * 2);
  float*  ssm  = (float*) carve((size_t)cT * cD * 4);
  bf16_t* n2bf = (bf16_t*)carve((size_t)cT * cD * 2);
  bf16_t* t1bf = (bf16_t*)carve((size_t)cT * cF * 2);
  bf16_t* hfbf = (bf16_t*)carve((size_t)cT * cD * 2);

  const dim3 blk(256);
  auto g1 = [](size_t n) { return dim3((unsigned)((n + 255) / 256)); };

  // ---- weight conversion (bf16, NT layouts; dt|B|C fused into one 768xD) ----
  k_to_bf16<<<g1((size_t)cV * cD), blk, 0, stream>>>(emb, emb_bf, (size_t)cV * cD);
  for (int l = 0; l < cL; ++l) {
    bf16_t* wc = WcatT + (size_t)l * c3N * cD;
    k_transpose_bf16<<<g1((size_t)cD * cN), blk, 0, stream>>>(
        W_dt + (size_t)l * cD * cN, wc,                          cD, cN);
    k_transpose_bf16<<<g1((size_t)cD * cN), blk, 0, stream>>>(
        W_B  + (size_t)l * cD * cN, wc + (size_t)cN * cD,        cD, cN);
    k_transpose_bf16<<<g1((size_t)cD * cN), blk, 0, stream>>>(
        W_C  + (size_t)l * cD * cN, wc + (size_t)2 * cN * cD,    cD, cN);
    k_transpose_bf16<<<g1((size_t)cN * cD), blk, 0, stream>>>(
        W_out + (size_t)l * cN * cD, WoutT + (size_t)l * cD * cN, cN, cD);
    k_transpose_bf16<<<g1((size_t)cD * cF), blk, 0, stream>>>(
        W1 + (size_t)l * cD * cF, W1T + (size_t)l * cF * cD, cD, cF);
    k_transpose_bf16<<<g1((size_t)cF * cD), blk, 0, stream>>>(
        W2 + (size_t)l * cF * cD, W2T + (size_t)l * cD * cF, cF, cD);
  }

  // ---- embedding gather ----
  k_gather<<<dim3(cT), blk, 0, stream>>>(ids, emb, x);

  // ---- layers ----
  const dim3 gProj(c3N / 128, cT / 64);  // (6, 64)
  const dim3 gOut (cD / 128, cT / 64);   // (8, 64)
  const dim3 gFfn1(cF / 128, cT / 64);   // (32, 64)
  const dim3 gFfn2(cD / 128, cT / 64);   // (8, 64)

  for (int l = 0; l < cL; ++l) {
    // LN1 -> h (f32 for D_skip path) + hbf (GEMM input)
    k_layernorm<3><<<dim3(cT), blk, 0, stream>>>(x, ln1s + (size_t)l * cD,
                                                 ln1b + (size_t)l * cD, h, hbf);
    // fused projections: [dt_raw | Bm | Cm]  (T x 768), one pass over h
    k_gemm_bf16_nt<EPI_NONE><<<gProj, blk, 0, stream>>>(
        hbf, WcatT + (size_t)l * c3N * cD, dbc, nullptr, cT, c3N, cD,
        nullptr, nullptr, nullptr, nullptr);
    // dt/decay/binp pointwise, then sequential scan producing bf16(h*C)
    k_ssm_elem<<<g1((size_t)cT * cN), blk, 0, stream>>>(
        dbc, b_dt + (size_t)l * cN, A_log + (size_t)l * cN);
    k_ssm_scan<<<dim3((cB * cN) / 256), blk, 0, stream>>>(dbc, prod);
    // out projection + D_skip + residual -> ssm_out
    k_gemm_bf16_nt<EPI_SSM><<<gOut, blk, 0, stream>>>(
        prod, WoutT + (size_t)l * cD * cN, ssm, nullptr, cT, cD, cN,
        nullptr, /*res=*/x, /*aux=*/h, /*dskip=*/Dskip + (size_t)l * cD);
    // LN2 -> n2 (bf16 only)
    k_layernorm<2><<<dim3(cT), blk, 0, stream>>>(ssm, ln2s + (size_t)l * cD,
                                                 ln2b + (size_t)l * cD, nullptr, n2bf);
    // FFN up: gelu(n2 @ W1 + b1) -> bf16
    k_gemm_bf16_nt<EPI_BIAS_GELU_BF><<<gFfn1, blk, 0, stream>>>(
        n2bf, W1T + (size_t)l * cF * cD, nullptr, t1bf, cT, cF, cD,
        b1 + (size_t)l * cF, nullptr, nullptr, nullptr);
    // FFN down + bias + residual(ssm_out) -> x (next layer input)
    k_gemm_bf16_nt<EPI_BIAS_RES><<<gFfn2, blk, 0, stream>>>(
        t1bf, W2T + (size_t)l * cD * cF, x, nullptr, cT, cD, cF,
        b2 + (size_t)l * cD, /*res=*/ssm, nullptr, nullptr);
  }

  // ---- final LN + tied-embedding logits ----
  k_layernorm<2><<<dim3(cT), blk, 0, stream>>>(x, lnfs, lnfb, nullptr, hfbf);
  const dim3 gLog(cV / 128, cT / 64);    // (250, 64)
  k_gemm_bf16_nt<EPI_NONE><<<gLog, blk, 0, stream>>>(
      hfbf, emb_bf, (float*)d_out, nullptr, cT, cV, cD,
      nullptr, nullptr, nullptr, nullptr);
}
